// GPTJBlockLast_71055938945068
// MI455X (gfx1250) — matrix operand order repaired and checked by reading, not compiled, not run on hardware
//
#include <hip/hip_runtime.h>
#include <hip/hip_bf16.h>

typedef __bf16 bf16;
typedef __attribute__((ext_vector_type(16))) __bf16 v16bf;
typedef __attribute__((ext_vector_type(8)))  float  v8f;
typedef __attribute__((ext_vector_type(4)))  unsigned int u32x4;
typedef __attribute__((ext_vector_type(8)))  int i32x8;
typedef __attribute__((ext_vector_type(4)))  int i32x4;

#define NE   4096
#define NI   16384
#define BB   32
#define HH   16
#define DD   256
#define RD_  64
#define KVL_ 1024

#if defined(__has_builtin)
#  if __has_builtin(__builtin_amdgcn_tensor_load_to_lds)
#    define HAVE_TDM 1
#  else
#    define HAVE_TDM 0
#  endif
#else
#  define HAVE_TDM 0
#endif

// ---------------------------------------------------------------- LayerNorm
__global__ void ln_kernel(const float* __restrict__ x, const float* __restrict__ w,
                          const float* __restrict__ bia, float* __restrict__ h,
                          bf16* __restrict__ hb) {
  __shared__ float red[256];
  int b = blockIdx.x, t = threadIdx.x;
  const float* xr = x + b * NE;
  float s = 0.f, s2 = 0.f;
  for (int i = t; i < NE; i += 256) { float v = xr[i]; s += v; s2 += v * v; }
  red[t] = s; __syncthreads();
  for (int k = 128; k; k >>= 1) { if (t < k) red[t] += red[t + k]; __syncthreads(); }
  float mean = red[0] / (float)NE; __syncthreads();
  red[t] = s2; __syncthreads();
  for (int k = 128; k; k >>= 1) { if (t < k) red[t] += red[t + k]; __syncthreads(); }
  float var  = red[0] / (float)NE - mean * mean;
  float rstd = rsqrtf(var + 1e-5f);
  for (int i = t; i < NE; i += 256) {
    float v = (xr[i] - mean) * rstd * w[i] + bia[i];
    h[b * NE + i]  = v;
    hb[b * NE + i] = (bf16)v;
  }
}

#if HAVE_TDM
// TDM: DMA a 32-row x 128-col fp32 tile of W (row stride N elems) into LDS.
// Pad: 1 DWORD after every 128 DWORDs -> LDS row stride = 129 floats
// (makes the two half-wave row groups bank-conflict-free).
__device__ __forceinline__ void tdm_load_tile(unsigned lds_off, const float* gptr, int N) {
  unsigned long long ga = (unsigned long long)(size_t)gptr;
  u32x4 g0;
  g0[0] = 1u;                                    // count=1, user descriptor
  g0[1] = lds_off;                               // lds_addr (bytes)
  g0[2] = (unsigned)ga;                          // global_addr[31:0]
  g0[3] = (unsigned)((ga >> 32) & 0x1FFFFFFu) | (2u << 30);  // addr[56:32] | type=2
  i32x8 g1;
  g1[0] = (int)((2u << 16) | (1u << 20) | (6u << 22)); // data_size=4B, pad_en, ival=128dw, amt=1dw
  g1[1] = (int)(128u << 16);                     // tensor_dim0 = 128
  g1[2] = (int)(32u << 16);                      // tensor_dim1 = 32
  g1[3] = (int)(128u << 16);                     // tile_dim0 = 128
  g1[4] = 32;                                    // tile_dim1 = 32, tile_dim2 = 0
  g1[5] = N;                                     // tensor_dim0_stride = N
  g1[6] = 0; g1[7] = 0;
  i32x4 z4 = {0, 0, 0, 0};
#if __clang_major__ >= 23
  i32x8 z8 = {0, 0, 0, 0, 0, 0, 0, 0};
  __builtin_amdgcn_tensor_load_to_lds(g0, g1, z4, z4, z8, 0);
#else
  __builtin_amdgcn_tensor_load_to_lds(g0, g1, z4, z4, 0);
#endif
}
#endif

// ------------------------------------------------- block GEMM: 32xK @ KxN
// Block = 256 threads (8 waves); each wave owns a 32x16 tile of a 32x128
// block tile. W tiles are TDM-DMA'd into double-buffered LDS; WMMA bf16.
template <int K>
__device__ __forceinline__ void gemm_tile(const bf16* __restrict__ A,
                                          const float* __restrict__ W,
                                          int N, int n0b, v8f& acc0, v8f& acc1) {
  __shared__ float sW[2][32 * 129];
  const int lane  = threadIdx.x & 31;
  const int half  = lane >> 4, lr = lane & 15;
  const int wtile = threadIdx.x >> 5;            // 0..7
  const int nloc  = wtile * 16 + lr;             // column within 128-wide tile

#if HAVE_TDM
  if (threadIdx.x < 32) {
    tdm_load_tile((unsigned)(size_t)&sW[0][0], W + n0b, N);
    __builtin_amdgcn_s_wait_tensorcnt(0);
  }
  __syncthreads();
#else
  for (int idx = threadIdx.x; idx < 32 * 128; idx += 256) {
    int r = idx >> 7, c = idx & 127;
    sW[0][r * 129 + c] = W[(long)r * N + n0b + c];
  }
  __syncthreads();
#endif

  for (int kb = 0; kb < K; kb += 32) {
    const int cur = (kb >> 5) & 1;
    const float* sB = sW[cur];
#if HAVE_TDM
    if (kb + 32 < K && threadIdx.x < 32)
      tdm_load_tile((unsigned)(size_t)&sW[cur ^ 1][0],
                    W + (long)(kb + 32) * N + n0b, N);
#endif
    v16bf a0, a1, bm;
#pragma unroll
    for (int e = 0; e < 16; ++e) {
      int kk = kb + 8 * half + (e < 8 ? e : e + 8);   // A 16x32 bf16 layout
      a0[e] = A[lr * K + kk];                          // rows 0..15
      a1[e] = A[(lr + 16) * K + kk];                   // rows 16..31
      bm[e] = (bf16)sB[(8 * half + (e < 8 ? e : e + 8)) * 129 + nloc];  // B 32x16 layout, from LDS
    }
    acc0 = __builtin_amdgcn_wmma_f32_16x16x32_bf16(false, a0, false, bm,
                                                   (short)0, acc0, false, false);
    acc1 = __builtin_amdgcn_wmma_f32_16x16x32_bf16(false, a1, false, bm,
                                                   (short)0, acc1, false, false);
    if (kb + 32 < K) {
#if HAVE_TDM
      if (threadIdx.x < 32) __builtin_amdgcn_s_wait_tensorcnt(0);
      __syncthreads();
#else
      __syncthreads();
      for (int idx = threadIdx.x; idx < 32 * 128; idx += 256) {
        int r = idx >> 7, c = idx & 127;
        sW[cur ^ 1][r * 129 + c] = W[(long)(kb + 32 + r) * N + n0b + c];
      }
      __syncthreads();
#endif
    }
  }
}

// ---------------------------------------------------------------- QKV GEMM
__global__ void qkv_kernel(const bf16* __restrict__ hb,
                           const float* __restrict__ wq, const float* __restrict__ wk,
                           const float* __restrict__ wv,
                           float* __restrict__ q, float* __restrict__ k,
                           float* __restrict__ v) {
  int mat = blockIdx.x >> 5;                    // 0..2 (uniform per block)
  int n0b = (blockIdx.x & 31) * 128;
  const float* W = (mat == 0) ? wq : (mat == 1 ? wk : wv);
  float* O = (mat == 0) ? q : (mat == 1 ? k : v);
  v8f a0 = {}, a1 = {};
  gemm_tile<NE>(hb, W, NE, n0b, a0, a1);
  int lane = threadIdx.x & 31, half = lane >> 4, lr = lane & 15;
  int col = n0b + (threadIdx.x >> 5) * 16 + lr;
#pragma unroll
  for (int r = 0; r < 8; ++r) {
    O[(r + 8 * half) * NE + col]      = a0[r];
    O[(r + 8 * half + 16) * NE + col] = a1[r];
  }
}

// ---------------------------------------------------------------- RoPE
__global__ void rope_kernel(float* __restrict__ q, float* __restrict__ k,
                            const int* __restrict__ pos) {
  int b = blockIdx.x, t = threadIdx.x;          // 512 threads: 16 heads x 32 pairs
  int h = t >> 5, i = t & 31;
  float p   = (float)pos[b];
  float inv = __expf(-((float)(2 * i) / (float)RD_) * 9.210340372f);
  float ang = p * inv, sn = __sinf(ang), cs = __cosf(ang);
  int base = b * NE + h * DD + 2 * i;
  float x1 = q[base], x2 = q[base + 1];
  q[base] = x1 * cs - x2 * sn; q[base + 1] = x2 * cs + x1 * sn;
  x1 = k[base]; x2 = k[base + 1];
  k[base] = x1 * cs - x2 * sn; k[base + 1] = x2 * cs + x1 * sn;
}

// ---------------------------------------------------------------- attention
__global__ void attn_kernel(const float* __restrict__ q, const float* __restrict__ kws,
                            const float* __restrict__ vws,
                            const float* __restrict__ kc, const float* __restrict__ vc,
                            const int* __restrict__ kidx, const int* __restrict__ vidx,
                            const int* __restrict__ nklo, const int* __restrict__ nvlo,
                            const float* __restrict__ mask, bf16* __restrict__ ctx) {
  __shared__ float qs[DD];
  __shared__ float sc[KVL_];
  __shared__ float red[256];
  __shared__ int nk[32], nv[32];
  int b = blockIdx.x >> 4, h = blockIdx.x & 15, t = threadIdx.x;
  qs[t] = q[b * NE + h * DD + t];
  if (t < 32) { nk[t] = nklo[t]; nv[t] = nvlo[t]; }
  __syncthreads();
  for (int l = t; l < KVL_; l += 256) {
    int src = kidx[b * KVL_ + l];
    const float* kp = kc + (long)src * NE + h * DD;
#pragma unroll
    for (int j = 0; j < 32; ++j)
      if (src == nk[j]) kp = kws + j * NE + h * DD;   // functional cache update
    float d = 0.f;
    for (int dd = 0; dd < DD; dd += 4) {
      float4 kv = *(const float4*)(kp + dd);
      d += qs[dd] * kv.x + qs[dd + 1] * kv.y + qs[dd + 2] * kv.z + qs[dd + 3] * kv.w;
    }
    sc[l] = d * 0.0625f + mask[b * KVL_ + l];
  }
  __syncthreads();
  float mx = -1e30f;
  for (int l = t; l < KVL_; l += 256) mx = fmaxf(mx, sc[l]);
  red[t] = mx; __syncthreads();
  for (int s = 128; s; s >>= 1) { if (t < s) red[t] = fmaxf(red[t], red[t + s]); __syncthreads(); }
  mx = red[0]; __syncthreads();
  float sm = 0.f;
  for (int l = t; l < KVL_; l += 256) { float e = __expf(sc[l] - mx); sc[l] = e; sm += e; }
  red[t] = sm; __syncthreads();
  for (int s = 128; s; s >>= 1) { if (t < s) red[t] += red[t + s]; __syncthreads(); }
  float inv = 1.f / red[0];
  __syncthreads();
  float acc = 0.f;
  for (int l = 0; l < KVL_; ++l) {
    int src = vidx[b * KVL_ + l];
    const float* vp = vc + (long)src * NE + h * DD;
#pragma unroll
    for (int j = 0; j < 32; ++j)
      if (src == nv[j]) vp = vws + j * NE + h * DD;
    acc += sc[l] * vp[t];
  }
  ctx[b * NE + h * DD + t] = (bf16)(acc * inv);
}

// ---------------------------------------------------------------- wo GEMM
__global__ void wo_kernel(const bf16* __restrict__ ctx, const float* __restrict__ wo,
                          float* __restrict__ ao) {
  int n0b = blockIdx.x * 128;
  v8f a0 = {}, a1 = {};
  gemm_tile<NE>(ctx, wo, NE, n0b, a0, a1);
  int lane = threadIdx.x & 31, half = lane >> 4, lr = lane & 15;
  int col = n0b + (threadIdx.x >> 5) * 16 + lr;
#pragma unroll
  for (int r = 0; r < 8; ++r) {
    ao[(r + 8 * half) * NE + col]      = a0[r];
    ao[(r + 8 * half + 16) * NE + col] = a1[r];
  }
}

__device__ __forceinline__ float gelu_tanh(float x) {
  return 0.5f * x * (1.0f + tanhf(0.7978845608f * (x + 0.044715f * x * x * x)));
}

// ---------------------------------------------------------------- fc_in + GELU
__global__ void fcin_kernel(const bf16* __restrict__ hb, const float* __restrict__ w,
                            const float* __restrict__ bia, bf16* __restrict__ m) {
  int n0b = blockIdx.x * 128;
  v8f a0 = {}, a1 = {};
  gemm_tile<NE>(hb, w, NI, n0b, a0, a1);
  int lane = threadIdx.x & 31, half = lane >> 4, lr = lane & 15;
  int col = n0b + (threadIdx.x >> 5) * 16 + lr;
  float bv = bia[col];
#pragma unroll
  for (int r = 0; r < 8; ++r) {
    m[(r + 8 * half) * NI + col]      = (bf16)gelu_tanh(a0[r] + bv);
    m[(r + 8 * half + 16) * NI + col] = (bf16)gelu_tanh(a1[r] + bv);
  }
}

// ------------------------------------------- fc_out + bias + attn + residual
__global__ void fcout_kernel(const bf16* __restrict__ m, const float* __restrict__ w,
                             const float* __restrict__ bia, const float* __restrict__ ao,
                             const float* __restrict__ resid, float* __restrict__ out) {
  int n0b = blockIdx.x * 128;
  v8f a0 = {}, a1 = {};
  gemm_tile<NI>(m, w, NE, n0b, a0, a1);
  int lane = threadIdx.x & 31, half = lane >> 4, lr = lane & 15;
  int col = n0b + (threadIdx.x >> 5) * 16 + lr;
  float bv = bia[col];
#pragma unroll
  for (int r = 0; r < 8; ++r) {
    int idx = (r + 8 * half) * NE + col;
    out[idx] = a0[r] + bv + ao[idx] + resid[idx];
    idx += 16 * NE;
    out[idx] = a1[r] + bv + ao[idx] + resid[idx];
  }
}

// ---------------------------------------------------------------- launch
extern "C" void kernel_launch(void* const* d_in, const int* in_sizes, int n_in,
                              void* d_out, int out_size, void* d_ws, size_t ws_size,
                              hipStream_t stream) {
  (void)in_sizes; (void)n_in; (void)out_size; (void)ws_size;
  const float* hs   = (const float*)d_in[0];
  const float* kc   = (const float*)d_in[1];
  const float* vc   = (const float*)d_in[2];
  const float* lnw  = (const float*)d_in[3];
  const float* lnb  = (const float*)d_in[4];
  const float* wq   = (const float*)d_in[5];
  const float* wk   = (const float*)d_in[6];
  const float* wv   = (const float*)d_in[7];
  const float* wo   = (const float*)d_in[8];
  const float* fiw  = (const float*)d_in[9];
  const float* fib  = (const float*)d_in[10];
  const float* fow  = (const float*)d_in[11];
  const float* fob  = (const float*)d_in[12];
  const float* mask = (const float*)d_in[13];
  const int* nkl    = (const int*)d_in[14];
  const int* nvl    = (const int*)d_in[15];
  const int* kidx   = (const int*)d_in[16];
  const int* vidx   = (const int*)d_in[17];
  const int* pos    = (const int*)d_in[18];
  float* out = (float*)d_out;

  float* wsf    = (float*)d_ws;
  float* h_f32  = wsf;                                   // 131072 f
  bf16*  h_bf   = (bf16*)(wsf + 131072);                 // 131072 bf16
  float* q_ws   = wsf + 131072 + 65536;                  // 131072 f
  float* k_ws   = q_ws + 131072;                         // 131072 f
  float* v_ws   = k_ws + 131072;                         // 131072 f
  float* ao_ws  = v_ws + 131072;                         // 131072 f
  bf16*  ctx_bf = (bf16*)(ao_ws + 131072);               // 131072 bf16
  bf16*  m_bf   = (bf16*)(ao_ws + 131072 + 65536);       // 524288 bf16

  ln_kernel   <<<BB, 256, 0, stream>>>(hs, lnw, lnb, h_f32, h_bf);
  qkv_kernel  <<<96, 256, 0, stream>>>(h_bf, wq, wk, wv, q_ws, k_ws, v_ws);
  rope_kernel <<<BB, 512, 0, stream>>>(q_ws, k_ws, pos);
  attn_kernel <<<BB * HH, 256, 0, stream>>>(q_ws, k_ws, v_ws, kc, vc,
                                            kidx, vidx, nkl, nvl, mask, ctx_bf);
  wo_kernel   <<<32, 256, 0, stream>>>(ctx_bf, wo, ao_ws);
  fcin_kernel <<<128, 256, 0, stream>>>(h_bf, fiw, fib, m_bf);
  fcout_kernel<<<32, 256, 0, stream>>>(m_bf, fow, fob, ao_ws, hs, out);
}
